// SelfAttention_2379411882220
// MI455X (gfx1250) — matrix operand order repaired and checked by reading, not compiled, not run on hardware
//
#include <hip/hip_runtime.h>

// ---------------------------------------------------------------------------
// SAGAN self-attention for gfx1250 (MI455X), flash-attention formulation.
// All GEMMs use v_wmma_f32_16x16x32_bf16 (wave32 WMMA, f32 accumulate).
// ---------------------------------------------------------------------------

typedef __attribute__((ext_vector_type(16))) __bf16 bf16x16;
typedef __attribute__((ext_vector_type(8)))  __bf16 bf16x8;
typedef __attribute__((ext_vector_type(4)))  __bf16 bf16x4;
typedef __attribute__((ext_vector_type(8)))  float  f32x8;

#define NPIX 4096   // H*W per batch
#define NB   4
#define CH   256
#define CK   32

// workspace byte offsets (all >=16KB aligned)
#define WS_XB   ((size_t)0)         // x as bf16           [NB*NPIX][CH]   8 MB
#define WS_F    ((size_t)8388608)   // f                   [NB*NPIX][CK]   1 MB
#define WS_G    ((size_t)9437184)   // g                   [NB*NPIX][CK]   1 MB
#define WS_HMT  ((size_t)10485760)  // hm transposed       [NB][CH][NPIX]  8 MB
#define WS_O    ((size_t)18874368)  // attention output    [NB*NPIX][CH]   8 MB
#define WS_WFT  ((size_t)27262976)  // Wf^T bf16           [CK][CH]        16 KB
#define WS_WGT  ((size_t)27279360)  // Wg^T bf16           [CK][CH]        16 KB
#define WS_WHT  ((size_t)27295744)  // Wh^T bf16           [CH][CH]        128 KB
#define WS_WVT  ((size_t)27426816)  // Wv^T bf16           [CH][CH]        128 KB
#define WS_END  ((size_t)27557888)

__device__ __forceinline__ f32x8 wmma_bf16(bf16x16 a, bf16x16 b, f32x8 c) {
  return __builtin_amdgcn_wmma_f32_16x16x32_bf16(false, a, false, b,
                                                 (short)0, c, false, false);
}

// A operand: 16x32 bf16 tile, row-major with leading dim ld.
// Lane L (L<16): row L, K = 0..15 ; lane L+16: row L, K = 16..31.
__device__ __forceinline__ bf16x16 load_a_frag(const __bf16* tile, int ld) {
  const int l = threadIdx.x & 31;
  const __bf16* p = tile + (size_t)(l & 15) * ld + ((l >> 4) << 4);
  bf16x8 lo = *(const bf16x8*)(p);
  bf16x8 hi = *(const bf16x8*)(p + 8);
  bf16x16 a;
#pragma unroll
  for (int i = 0; i < 8; ++i) { a[i] = lo[i]; a[i + 8] = hi[i]; }
  return a;
}

// B operand: 32x16 bf16 tile given as B^T (row n holds column n of B,
// contiguous in K) with leading dim ld. Lane n: col n&15, K = (n>>4)*16 .. +15.
__device__ __forceinline__ bf16x16 load_b_frag(const __bf16* tileT, int ld) {
  const int l = threadIdx.x & 31;
  return *(const bf16x16*)(tileT + (size_t)(l & 15) * ld + ((l >> 4) << 4));
}

// --------------------------- prep kernels ----------------------------------

// transpose + convert all four weight matrices to bf16
__global__ void prep_weights_kernel(const float* Wf, const float* Wg,
                                    const float* Wh, const float* Wv,
                                    __bf16* WfT, __bf16* WgT,
                                    __bf16* WhT, __bf16* WvT) {
  int idx = blockIdx.x * blockDim.x + threadIdx.x;   // < 147456, exact grid
  if (idx < 8192) {                                  // WfT[o][i] = Wf[i][o]
    int o = idx >> 8, i = idx & 255;
    WfT[o * 256 + i] = (__bf16)Wf[i * 32 + o];
  } else if (idx < 16384) {
    int j = idx - 8192;  int o = j >> 8, i = j & 255;
    WgT[o * 256 + i] = (__bf16)Wg[i * 32 + o];
  } else if (idx < 81920) {
    int j = idx - 16384; int o = j >> 8, i = j & 255;
    WhT[o * 256 + i] = (__bf16)Wh[i * 256 + o];
  } else {
    int j = idx - 81920; int o = j >> 8, i = j & 255;
    WvT[o * 256 + i] = (__bf16)Wv[i * 256 + o];
  }
}

// convert x to bf16, 4 elements per thread (b128 loads)
__global__ void prep_x_kernel(const float* x, __bf16* xb) {
  int i = blockIdx.x * blockDim.x + threadIdx.x;     // < 1048576, exact grid
  float4 v = ((const float4*)x)[i];
  bf16x4 p;
  p[0] = (__bf16)v.x; p[1] = (__bf16)v.y; p[2] = (__bf16)v.z; p[3] = (__bf16)v.w;
  *(bf16x4*)(xb + (size_t)i * 4) = p;
}

// --------------------------- projections -----------------------------------
// One wave per 16 pixels: f (2 tiles), g (2 tiles), hm (16 tiles, stored
// transposed as hmT[b][c][n]); K loop over C=256 in 8 WMMA steps.
__global__ void proj_kernel(const __bf16* xb,
                            const __bf16* WfT, const __bf16* WgT, const __bf16* WhT,
                            const float* bf_, const float* bg_, const float* bh_,
                            __bf16* f, __bf16* g, __bf16* hmT) {
  const int wave = blockIdx.x * 8 + (threadIdx.x >> 5);
  const int l    = threadIdx.x & 31;
  const int pb   = wave * 16;                 // flat pixel base
  const int lo16 = l & 15, hi8 = (l >> 4) * 8;

  // x A-fragments for all 8 K-steps (reused by all 20 output tiles)
  bf16x16 ax[8];
  const __bf16* xrow = xb + (size_t)pb * CH;
#pragma unroll
  for (int k = 0; k < 8; ++k) ax[k] = load_a_frag(xrow + k * 32, CH);

  // ---- f and g: Cout = 32 -> 2 tiles each, strided bf16 stores
#pragma unroll
  for (int t = 0; t < 2; ++t) {
    f32x8 af = {0,0,0,0,0,0,0,0}, ag = {0,0,0,0,0,0,0,0};
#pragma unroll
    for (int k = 0; k < 8; ++k) {
      af = wmma_bf16(ax[k], load_b_frag(WfT + (size_t)(t * 16) * CH + k * 32, CH), af);
      ag = wmma_bf16(ax[k], load_b_frag(WgT + (size_t)(t * 16) * CH + k * 32, CH), ag);
    }
    const int cout = t * 16 + lo16;
    const float biasf = bf_[cout], biasg = bg_[cout];
#pragma unroll
    for (int v = 0; v < 8; ++v) {
      f[(size_t)(pb + hi8 + v) * CK + cout] = (__bf16)(af[v] + biasf);
      g[(size_t)(pb + hi8 + v) * CK + cout] = (__bf16)(ag[v] + biasg);
    }
  }

  // ---- hm: 16 tiles, stored transposed (contiguous 16B stores per lane)
  const int b   = pb >> 12;        // batch
  const int pin = pb & (NPIX - 1); // pixel within batch
#pragma unroll
  for (int t = 0; t < 16; ++t) {
    f32x8 acc = {0,0,0,0,0,0,0,0};
#pragma unroll
    for (int k = 0; k < 8; ++k)
      acc = wmma_bf16(ax[k], load_b_frag(WhT + (size_t)(t * 16) * CH + k * 32, CH), acc);
    const int cout = t * 16 + lo16;
    const float bias = bh_[cout];
    bf16x8 pk;
#pragma unroll
    for (int v = 0; v < 8; ++v) pk[v] = (__bf16)(acc[v] + bias);
    *(bf16x8*)(hmT + ((size_t)b * CH + cout) * NPIX + pin + hi8) = pk;
  }
}

// --------------------------- flash attention --------------------------------
// One wave per 16 queries. Key loop in chunks of 32:
//   s^T = f_tile(16k x 32ck) x g^T(32ck x 16q)  -> keys in VGPRs, queries in lanes
//   per-lane online softmax (one shfl_xor(16) merges the two half-lanes/query)
//   beta staged through LDS -> B operand; o^T accumulated over 16 channel tiles.
__global__ void attn_kernel(const __bf16* f, const __bf16* g,
                            const __bf16* hmT, __bf16* o) {
  __shared__ __align__(64) __bf16 betaLds[8][16][32];   // per-wave 1 KB

  const int wv   = threadIdx.x >> 5;
  const int wave = blockIdx.x * 8 + wv;
  const int l    = threadIdx.x & 31;
  const int b    = wave >> 8;              // 256 query tiles per batch
  const int qt   = wave & 255;
  const int qbase = qt * 16;
  const int lo16 = l & 15, hi8 = (l >> 4) * 8;

  const __bf16* fB = f   + (size_t)b * NPIX * CK;
  const __bf16* gB = g   + (size_t)b * NPIX * CK;
  const __bf16* hB = hmT + (size_t)b * CH * NPIX;

  // g^T B-operand for this wave's 16 queries (constant over the key loop)
  const bf16x16 gfrag = load_b_frag(gB + (size_t)qbase * CK, CK);

  f32x8 acc[16];                 // o^T accumulators: 16 channel tiles x 8 f32
#pragma unroll
  for (int t = 0; t < 16; ++t)
#pragma unroll
    for (int i = 0; i < 8; ++i) acc[t][i] = 0.0f;

  float m_run = -3.0e38f, l_run = 0.0f;
  const f32x8 zero = {0,0,0,0,0,0,0,0};

  for (int kc = 0; kc < NPIX; kc += 32) {
    // scores for 32 keys x 16 queries (two s^T tiles), full Ck=32 in one WMMA
    bf16x16 fa0 = load_a_frag(fB + (size_t)kc * CK, CK);
    bf16x16 fa1 = load_a_frag(fB + (size_t)(kc + 16) * CK, CK);
    f32x8 s0 = wmma_bf16(fa0, gfrag, zero);
    f32x8 s1 = wmma_bf16(fa1, gfrag, zero);

    // online softmax (per lane; partner lane l^16 holds same query)
    float cm = -3.0e38f;
#pragma unroll
    for (int i = 0; i < 8; ++i) cm = fmaxf(cm, fmaxf(s0[i], s1[i]));
    cm = fmaxf(cm, __shfl_xor(cm, 16, 32));
    const float mnew  = fmaxf(m_run, cm);
    const float scale = __expf(m_run - mnew);
    m_run = mnew;

    float rs = 0.0f;
    bf16x8 e0, e1;
#pragma unroll
    for (int i = 0; i < 8; ++i) {
      float t0 = __expf(s0[i] - mnew); rs += t0; e0[i] = (__bf16)t0;
      float t1 = __expf(s1[i] - mnew); rs += t1; e1[i] = (__bf16)t1;
    }
    rs += __shfl_xor(rs, 16, 32);
    l_run = l_run * scale + rs;

#pragma unroll
    for (int t = 0; t < 16; ++t)
#pragma unroll
      for (int i = 0; i < 8; ++i) acc[t][i] *= scale;

    // stage beta^T as beta[q][key] (contiguous 16B per lane), re-read as B op
    *(bf16x8*)&betaLds[wv][lo16][hi8]      = e0;   // keys kc+ 0..15
    *(bf16x8*)&betaLds[wv][lo16][16 + hi8] = e1;   // keys kc+16..31
    asm volatile("s_wait_dscnt 0x0" ::: "memory");  // DS ops in-order per wave
    const bf16x16 bfrag = *(const bf16x16*)&betaLds[wv][lo16][(l >> 4) << 4];

    // o^T += hmT_tile(16c x 32k) x beta(32k x 16q), 16 channel tiles
    const __bf16* hK = hB + kc;
#pragma unroll
    for (int t = 0; t < 16; ++t) {
      bf16x16 ha = load_a_frag(hK + (size_t)(t * 16) * NPIX, NPIX);
      acc[t] = wmma_bf16(ha, bfrag, acc[t]);
    }
    __builtin_prefetch(hB + kc + 32, 0, 1);   // next key chunk of hmT
  }

  // normalize and store o row-major bf16 (contiguous 16B per lane)
  const float inv = 1.0f / l_run;
  const int q = qbase + lo16;
#pragma unroll
  for (int t = 0; t < 16; ++t) {
    bf16x8 pk;
#pragma unroll
    for (int i = 0; i < 8; ++i) pk[i] = (__bf16)(acc[t][i] * inv);
    *(bf16x8*)(o + ((size_t)b * NPIX + q) * CH + t * 16 + hi8) = pk;
  }
}

// --------------------------- output projection ------------------------------
// out = x + o @ Wv + bv  (16 pixels per wave, 16 Cout tiles, f32 epilogue)
__global__ void outproj_kernel(const __bf16* ob, const __bf16* WvT,
                               const float* bv_, const float* x, float* out) {
  const int wave = blockIdx.x * 8 + (threadIdx.x >> 5);
  const int l    = threadIdx.x & 31;
  const int pb   = wave * 16;
  const int lo16 = l & 15, hi8 = (l >> 4) * 8;

  bf16x16 ax[8];
  const __bf16* orow = ob + (size_t)pb * CH;
#pragma unroll
  for (int k = 0; k < 8; ++k) ax[k] = load_a_frag(orow + k * 32, CH);

#pragma unroll
  for (int t = 0; t < 16; ++t) {
    f32x8 acc = {0,0,0,0,0,0,0,0};
#pragma unroll
    for (int k = 0; k < 8; ++k)
      acc = wmma_bf16(ax[k], load_b_frag(WvT + (size_t)(t * 16) * CH + k * 32, CH), acc);
    const int cout = t * 16 + lo16;
    const float bias = bv_[cout];
#pragma unroll
    for (int v = 0; v < 8; ++v) {
      const size_t idx = (size_t)(pb + hi8 + v) * CH + cout;
      out[idx] = acc[v] + bias + x[idx];
    }
  }
}

// --------------------------- host launcher ----------------------------------
extern "C" void kernel_launch(void* const* d_in, const int* in_sizes, int n_in,
                              void* d_out, int out_size, void* d_ws, size_t ws_size,
                              hipStream_t stream) {
  (void)in_sizes; (void)n_in; (void)out_size;
  if (ws_size < WS_END) return;   // need ~26.3 MB scratch

  const float* x  = (const float*)d_in[0];
  const float* Wf = (const float*)d_in[1];
  const float* bf = (const float*)d_in[2];
  const float* Wg = (const float*)d_in[3];
  const float* bg = (const float*)d_in[4];
  const float* Wh = (const float*)d_in[5];
  const float* bh = (const float*)d_in[6];
  const float* Wv = (const float*)d_in[7];
  const float* bv = (const float*)d_in[8];
  float* out = (float*)d_out;

  char* ws = (char*)d_ws;
  __bf16* xb  = (__bf16*)(ws + WS_XB);
  __bf16* f   = (__bf16*)(ws + WS_F);
  __bf16* g   = (__bf16*)(ws + WS_G);
  __bf16* hmT = (__bf16*)(ws + WS_HMT);
  __bf16* ob  = (__bf16*)(ws + WS_O);
  __bf16* WfT = (__bf16*)(ws + WS_WFT);
  __bf16* WgT = (__bf16*)(ws + WS_WGT);
  __bf16* WhT = (__bf16*)(ws + WS_WHT);
  __bf16* WvT = (__bf16*)(ws + WS_WVT);

  prep_weights_kernel<<<576, 256, 0, stream>>>(Wf, Wg, Wh, Wv, WfT, WgT, WhT, WvT);
  prep_x_kernel<<<4096, 256, 0, stream>>>(x, xb);
  // 4*4096 pixels / 16 per wave / 8 waves per block = 128 blocks
  proj_kernel<<<128, 256, 0, stream>>>(xb, WfT, WgT, WhT, bf, bg, bh, f, g, hmT);
  attn_kernel<<<128, 256, 0, stream>>>(f, g, hmT, ob);
  outproj_kernel<<<128, 256, 0, stream>>>(ob, WvT, bv, x, out);
}
